// Semantic_5454608466441
// MI455X (gfx1250) — matrix operand order repaired and checked by reading, not compiled, not run on hardware
//
#include <hip/hip_runtime.h>

// Shapes fixed by the reference
#define STATE_DIM 512
#define PROJ      300
#define PROJ_PAD  304          // 19 tiles of 16
#define NT        19
#define NODE      32
#define BATSEQ    320
#define EDGES     1024
#define NODE_ROWS (BATSEQ * NODE)      // 10240
#define EDGE_ROWS (BATSEQ * EDGES)     // 327680
#define MT_NODE   (NODE_ROWS / 16)     // 640
#define MT_EDGE   (EDGE_ROWS / 16)     // 20480
// losses = sum / (batseq * PROJ) / batch ; batch = 320/10 = 32
#define INV_SCALE (1.0f / 3072000.0f)

typedef __attribute__((ext_vector_type(16))) _Float16 v16h;
typedef __attribute__((ext_vector_type(8)))  float    v8f;
typedef __attribute__((ext_vector_type(4)))  float    v4f;

// ---------------------------------------------------------------------------
// Load 16x32 f16 A-fragment for one lane from an f32 row-major matrix.
// Lane (l, hi): row = l, contiguous k range [kb+16*hi, +16).
// ---------------------------------------------------------------------------
__device__ __forceinline__ v16h load_a_frag(const float* __restrict__ row,
                                            int kb, int hi) {
    const v4f* p0 = (const v4f*)(row + kb + 16 * hi);
    const v4f* p1 = (const v4f*)(row + kb + 16 * hi + 8);
    v4f a0 = p0[0], a1 = p0[1];
    v4f a2 = p1[0], a3 = p1[1];
    v16h r;
#pragma unroll
    for (int t = 0; t < 4; ++t) {
        r[t]      = (_Float16)a0[t];
        r[4 + t]  = (_Float16)a1[t];
        r[8 + t]  = (_Float16)a2[t];
        r[12 + t] = (_Float16)a3[t];
    }
    return r;
}

// ---------------------------------------------------------------------------
// Kernel 1: convert W1/W2/W3 -> f16 padded [3][304][512], biases -> [3][304]
// ---------------------------------------------------------------------------
__global__ void __launch_bounds__(256) k_prep(
    const float* __restrict__ W1, const float* __restrict__ b1,
    const float* __restrict__ W2, const float* __restrict__ b2,
    const float* __restrict__ W3, const float* __restrict__ b3,
    _Float16* __restrict__ Wh, float* __restrict__ bp)
{
    int t = blockIdx.x * blockDim.x + threadIdx.x;
    const int total = 3 * PROJ_PAD * STATE_DIM;
    if (t < total) {
        int w = t / (PROJ_PAD * STATE_DIM);
        int r = t - w * (PROJ_PAD * STATE_DIM);
        int p = r / STATE_DIM;
        int k = r - p * STATE_DIM;
        const float* W = (w == 0) ? W1 : (w == 1) ? W2 : W3;
        float v = (p < PROJ) ? W[p * STATE_DIM + k] : 0.0f;
        Wh[t] = (_Float16)v;
    }
    if (t < 3 * PROJ_PAD) {
        int w = t / PROJ_PAD;
        int p = t - w * PROJ_PAD;
        const float* B = (w == 0) ? b1 : (w == 1) ? b2 : b3;
        bp[t] = (p < PROJ) ? B[p] : 0.0f;
    }
}

// ---------------------------------------------------------------------------
// Kernel 2: f1 = nodes @ W1^T + b1, f3 = nodes @ W3^T + b3  (f32 [10240][304])
// One wave per 16-row strip; all 19 N-tiles kept in registers (A read once).
// grid = (MT_NODE, 2), block = 32
// ---------------------------------------------------------------------------
__global__ void __launch_bounds__(32) k_proj(
    const float* __restrict__ nodes, const _Float16* __restrict__ Wh,
    const float* __restrict__ bp, float* __restrict__ f1,
    float* __restrict__ f3)
{
    const int lane = threadIdx.x;
    const int l = lane & 15, hi = lane >> 4;
    const int mt = blockIdx.x;
    const int which = blockIdx.y;                     // 0 -> (W1,f1), 1 -> (W3,f3)
    const size_t wsel = (which == 0) ? 0 : 2;
    const _Float16* __restrict__ W = Wh + wsel * (size_t)PROJ_PAD * STATE_DIM;
    const float* __restrict__ bias = bp + wsel * PROJ_PAD;
    float* __restrict__ out = (which == 0) ? f1 : f3;

    const int rowBase = mt * 16;
    const float* __restrict__ arow = nodes + (size_t)(rowBase + l) * STATE_DIM;

    v8f acc[NT];
#pragma unroll
    for (int n = 0; n < NT; ++n)
#pragma unroll
        for (int r = 0; r < 8; ++r) acc[n][r] = 0.0f;

    for (int kt = 0; kt < STATE_DIM / 32; ++kt) {
        const int kb = kt * 32;
        v16h a = load_a_frag(arow, kb, hi);
#pragma unroll
        for (int n = 0; n < NT; ++n) {
            v16h bf = *(const v16h*)(W + (size_t)(n * 16 + l) * STATE_DIM
                                       + kb + 16 * hi);
            acc[n] = __builtin_amdgcn_wmma_f32_16x16x32_f16(
                false, a, false, bf, (short)0, acc[n], false, false);
        }
    }

#pragma unroll
    for (int n = 0; n < NT; ++n) {
        const int p = n * 16 + l;
        const float bv = bias[p];
#pragma unroll
        for (int r = 0; r < 8; ++r) {
            const int m = r + 8 * hi;
            out[(size_t)(rowBase + m) * PROJ_PAD + p] = acc[n][r] + bv;
        }
    }
}

// ---------------------------------------------------------------------------
// Kernel 3: fused f2 GEMM + loss.  Each wave computes a 16x304 f2 strip in
// registers, then accumulates sum((f3[j] - f2 - (f1[i]+b2))^2) -> partial.
// grid = MT_EDGE, block = 32.  f2 never touches memory.
// ---------------------------------------------------------------------------
__global__ void __launch_bounds__(32) k_loss(
    const float* __restrict__ edges, const _Float16* __restrict__ Wh,
    const float* __restrict__ bp, const float* __restrict__ f1,
    const float* __restrict__ f3, float* __restrict__ partials)
{
    const int lane = threadIdx.x;
    const int l = lane & 15, hi = lane >> 4;
    const int mt = blockIdx.x;
    const int e0 = mt * 16;                 // global edge row of this tile
    const int b = e0 / EDGES;
    const int eloc = e0 - b * EDGES;
    const int i = eloc >> 5;                // fixed node i for this tile
    const int j0 = eloc & 31;               // 0 or 16

    const _Float16* __restrict__ W2h = Wh + (size_t)PROJ_PAD * STATE_DIM;
    const float* __restrict__ b2p = bp + PROJ_PAD;
    const float* __restrict__ arow = edges + (size_t)(e0 + l) * STATE_DIM;

    v8f acc[NT];
#pragma unroll
    for (int n = 0; n < NT; ++n)
#pragma unroll
        for (int r = 0; r < 8; ++r) acc[n][r] = 0.0f;

    for (int kt = 0; kt < STATE_DIM / 32; ++kt) {
        const int kb = kt * 32;
        v16h a = load_a_frag(arow, kb, hi);
#pragma unroll
        for (int n = 0; n < NT; ++n) {
            v16h bf = *(const v16h*)(W2h + (size_t)(n * 16 + l) * STATE_DIM
                                         + kb + 16 * hi);
            acc[n] = __builtin_amdgcn_wmma_f32_16x16x32_f16(
                false, a, false, bf, (short)0, acc[n], false, false);
        }
    }

    // Epilogue: diff = f3[b, j0+m, p] - (acc + b2[p]) - f1[b, i, p]
    const float* __restrict__ f1row  = f1 + (size_t)(b * NODE + i) * PROJ_PAD;
    const float* __restrict__ f3base = f3 + (size_t)(b * NODE + j0) * PROJ_PAD;

    float local = 0.0f;
#pragma unroll
    for (int n = 0; n < NT; ++n) {
        const int p = n * 16 + l;
        const float c = f1row[p] + b2p[p];
#pragma unroll
        for (int r = 0; r < 8; ++r) {
            const int m = r + 8 * hi;
            const float d = f3base[(size_t)m * PROJ_PAD + p] - acc[n][r] - c;
            local = fmaf(d, d, local);
        }
    }

    // wave32 reduction (deterministic), one partial per wave
#pragma unroll
    for (int off = 16; off > 0; off >>= 1)
        local += __shfl_down(local, off, 32);
    if (lane == 0) partials[mt] = local;
}

// ---------------------------------------------------------------------------
// Kernel 4: deterministic final reduction of MT_EDGE partials -> scalar loss
// ---------------------------------------------------------------------------
__global__ void __launch_bounds__(256) k_reduce(
    const float* __restrict__ partials, float* __restrict__ out)
{
    __shared__ float sm[256];
    float s = 0.0f;
    for (int idx = threadIdx.x; idx < MT_EDGE; idx += 256) s += partials[idx];
    sm[threadIdx.x] = s;
    __syncthreads();
    for (int off = 128; off > 0; off >>= 1) {
        if (threadIdx.x < off) sm[threadIdx.x] += sm[threadIdx.x + off];
        __syncthreads();
    }
    if (threadIdx.x == 0) out[0] = sm[0] * INV_SCALE;
}

// ---------------------------------------------------------------------------
// Host launch.  Input order: nodes(0) edges(1) adj(2,unused) true_lengths(3,
// unused) seq_size(4,unused; batch=32 folded into INV_SCALE) W1 b1 W2 b2 W3 b3
// ---------------------------------------------------------------------------
extern "C" void kernel_launch(void* const* d_in, const int* in_sizes, int n_in,
                              void* d_out, int out_size, void* d_ws, size_t ws_size,
                              hipStream_t stream) {
    (void)in_sizes; (void)n_in; (void)out_size; (void)ws_size;
    const float* nodes = (const float*)d_in[0];
    const float* edges = (const float*)d_in[1];
    const float* W1 = (const float*)d_in[5];
    const float* b1 = (const float*)d_in[6];
    const float* W2 = (const float*)d_in[7];
    const float* b2 = (const float*)d_in[8];
    const float* W3 = (const float*)d_in[9];
    const float* b3 = (const float*)d_in[10];
    float* out = (float*)d_out;

    // Workspace layout (bytes)
    char* ws = (char*)d_ws;
    size_t off = 0;
    _Float16* Wh = (_Float16*)(ws + off);               // 3*304*512 f16
    off += (size_t)3 * PROJ_PAD * STATE_DIM * sizeof(_Float16);
    float* bp = (float*)(ws + off);                     // 3*304 f32
    off += (size_t)3 * PROJ_PAD * sizeof(float);
    off = (off + 255) & ~(size_t)255;
    float* f1 = (float*)(ws + off);                     // 10240*304 f32
    off += (size_t)NODE_ROWS * PROJ_PAD * sizeof(float);
    float* f3 = (float*)(ws + off);                     // 10240*304 f32
    off += (size_t)NODE_ROWS * PROJ_PAD * sizeof(float);
    float* partials = (float*)(ws + off);               // 20480 f32

    {
        const int total = 3 * PROJ_PAD * STATE_DIM;
        k_prep<<<(total + 255) / 256, 256, 0, stream>>>(W1, b1, W2, b2, W3, b3,
                                                        Wh, bp);
    }
    {
        dim3 grid(MT_NODE, 2);
        k_proj<<<grid, 32, 0, stream>>>(nodes, Wh, bp, f1, f3);
    }
    k_loss<<<MT_EDGE, 32, 0, stream>>>(edges, Wh, bp, f1, f3, partials);
    k_reduce<<<1, 256, 0, stream>>>(partials, out);
}
